// Router_52888227283719
// MI455X (gfx1250) — matrix operand order repaired and checked by reading, not compiled, not run on hardware
//
#include <hip/hip_runtime.h>
#include <hip/hip_bf16.h>

typedef __attribute__((ext_vector_type(16))) __bf16       v16bf;
typedef __attribute__((ext_vector_type(8)))  float        v8f;
typedef __attribute__((ext_vector_type(4)))  unsigned int u32x4;
typedef __attribute__((ext_vector_type(8)))  int          i32x8;
typedef __attribute__((ext_vector_type(4)))  int          i32x4;

#define D_MODEL 2048
#define NEXP    64
#define BM      128
#define WPITCH  2064                      // bf16 elems per W row (4128B: 8-bank lane stride, 32B aligned)
#define LDS_W_BYTES   (NEXP * WPITCH * 2) // 264192
#define LDS_LOG_OFF   LDS_W_BYTES
#define LDS_CNT_OFF   (LDS_LOG_OFF + BM * NEXP * 4)
#define LDS_TOTAL     (LDS_CNT_OFF + NEXP * 4)   // ~290 KB of the 320 KB WGP LDS

// ---------------- W f32 -> bf16 into padded-pitch global staging buffer -----
__global__ void Router_cvtw_kernel(const float* __restrict__ W,
                                   __bf16* __restrict__ Wb, int n) {
  int i = blockIdx.x * blockDim.x + threadIdx.x;
  if (i < n) {
    int e = i >> 11;          // / D_MODEL
    int k = i & (D_MODEL - 1);
    Wb[e * WPITCH + k] = (__bf16)W[i];
  }
}

__global__ void Router_zero_kernel(float* __restrict__ acc) {
  int i = threadIdx.x;
  if (i < 2 * NEXP) acc[i] = 0.0f;
}

// ---------------- main router: TDM-staged B + bf16 WMMA GEMM + softmax/top-2
__global__ __launch_bounds__(256)
void Router_main_kernel(const float* __restrict__ x,
                        const __bf16* __restrict__ Wb,
                        const float* __restrict__ bias,
                        float* __restrict__ out_idx,
                        float* __restrict__ out_w,
                        float* __restrict__ out_logits,
                        float* __restrict__ g_cnt,
                        float* __restrict__ g_imp) {
  extern __shared__ char smem[];
  __bf16* s_w      = (__bf16*)smem;                  // 64 x 2064 bf16, pitch 4128B
  float*  s_logits = (float*)(smem + LDS_LOG_OFF);   // 128 x 64
  float*  s_cnt    = (float*)(smem + LDS_CNT_OFF);   // 64

  const int tid    = threadIdx.x;
  const int wave   = tid >> 5;
  const int lane   = tid & 31;
  const int lane15 = lane & 15;
  const int laneHi = lane >> 4;

  if (tid < NEXP) s_cnt[tid] = 0.0f;

  // ---- TDM: stage entire bf16 W (2064 x 64 tile, 2B elems) into LDS --------
  if (tid == 0) {
    unsigned           lds_a = (unsigned)(unsigned long long)s_w;
    unsigned long long ga    = (unsigned long long)Wb;
    u32x4 g0;
    g0[0] = 1u;                                        // count=1, load, no gather
    g0[1] = lds_a;                                     // lds_addr
    g0[2] = (unsigned)ga;                              // global_addr[31:0]
    g0[3] = (unsigned)((ga >> 32) & 0x01FFFFFFu) | (2u << 30);  // addr[56:32] | type=2
    i32x8 g1;
    g1[0] = (1 << 16);                                 // data_size=1 (2B), no pad/iter/multicast
    g1[1] = (int)((unsigned)WPITCH << 16);             // tensor_dim0[15:0] @ [63:48]
    g1[2] = (int)((unsigned)NEXP << 16);               // tensor_dim0 hi=0, tensor_dim1=64 @ [95:80]
    g1[3] = (int)((unsigned)WPITCH << 16);             // tile_dim0=2064 @ [127:112]
    g1[4] = NEXP;                                      // tile_dim1=64, tile_dim2=0
    g1[5] = WPITCH;                                    // tensor_dim0_stride[31:0]
    g1[6] = 0;
    g1[7] = 0;
    i32x4 z4 = {0, 0, 0, 0};
    i32x8 z8 = {0, 0, 0, 0, 0, 0, 0, 0};
    __builtin_amdgcn_tensor_load_to_lds(g0, g1, z4, z4, z8, 0);
  }
  __builtin_amdgcn_s_wait_tensorcnt(0);   // no-op for waves with TENSORcnt==0
  __syncthreads();                        // publish W tile to all waves

  // ---- GEMM: each wave computes 16 rows x 64 experts ------------------------
  const int  rbase_l = wave * 16;
  const long rowA    = (long)blockIdx.x * BM + rbase_l + lane15;
  const float* __restrict__ xrow = x + rowA * D_MODEL;

  v8f acc[4] = {};

  for (int k0 = 0; k0 < D_MODEL; k0 += 32) {
    const int kA = k0 + laneHi * 16;  // A layout: lo lanes K0-15, hi lanes K16-31
    float4 a0 = *(const float4*)(xrow + kA);
    float4 a1 = *(const float4*)(xrow + kA + 4);
    float4 a2 = *(const float4*)(xrow + kA + 8);
    float4 a3 = *(const float4*)(xrow + kA + 12);
    __builtin_prefetch(xrow + kA + 32, 0, 3);   // next K-chunk

    v16bf af;
    af[0]  = (__bf16)a0.x; af[1]  = (__bf16)a0.y; af[2]  = (__bf16)a0.z; af[3]  = (__bf16)a0.w;
    af[4]  = (__bf16)a1.x; af[5]  = (__bf16)a1.y; af[6]  = (__bf16)a1.z; af[7]  = (__bf16)a1.w;
    af[8]  = (__bf16)a2.x; af[9]  = (__bf16)a2.y; af[10] = (__bf16)a2.z; af[11] = (__bf16)a2.w;
    af[12] = (__bf16)a3.x; af[13] = (__bf16)a3.y; af[14] = (__bf16)a3.z; af[15] = (__bf16)a3.w;

    const int kB = k0 + laneHi * 16;  // B layout: lane = expert column, K packed along VGPRs
#pragma unroll
    for (int t = 0; t < 4; ++t) {
      const __bf16* wp = s_w + (size_t)(t * 16 + lane15) * WPITCH + kB;  // 32B-aligned LDS
      v16bf bf = *(const v16bf*)wp;   // 2x ds_load_b128, conflict-free pitch
      acc[t] = __builtin_amdgcn_wmma_f32_16x16x32_bf16(
          false, af, false, bf, (short)0, acc[t], false, false);
    }
  }

  // ---- bias add + store logits (global + LDS) -------------------------------
  // C layout: VGPR i, lanes 0-15 -> M=i, N=lane; lanes 16-31 -> M=i+8, N=lane-16
#pragma unroll
  for (int t = 0; t < 4; ++t) {
    const int   e  = t * 16 + lane15;
    const float be = bias[e];
#pragma unroll
    for (int i = 0; i < 8; ++i) {
      const int   lr = rbase_l + laneHi * 8 + i;
      const float v  = acc[t][i] + be;
      s_logits[lr * NEXP + e] = v;
      out_logits[((long)blockIdx.x * BM + lr) * NEXP + e] = v;
    }
  }
  __syncthreads();

  // ---- softmax + top-2 per token (1 thread per row) -------------------------
  if (tid < BM) {
    float* rowp = s_logits + tid * NEXP;
    float  m = rowp[0];
#pragma unroll
    for (int e = 1; e < NEXP; ++e) m = fmaxf(m, rowp[e]);
    float sum = 0.0f, b1 = -1e30f, b2 = -1e30f;
    int   i1 = 0, i2 = 0;
#pragma unroll
    for (int e = 0; e < NEXP; ++e) {
      float p = __expf(rowp[e] - m);
      rowp[e] = p;
      sum += p;
      if (p > b1)      { b2 = b1; i2 = i1; b1 = p; i1 = e; }
      else if (p > b2) { b2 = p; i2 = e; }
    }
    const float inv = 1.0f / sum;
#pragma unroll
    for (int e = 0; e < NEXP; ++e) rowp[e] *= inv;  // routing probs for importance sums

    const long  token = (long)blockIdx.x * BM + tid;
    const float wsum  = 1.0f / (b1 + b2);
    out_idx[token * 2 + 0] = (float)i1;
    out_idx[token * 2 + 1] = (float)i2;
    out_w[token * 2 + 0]   = b1 * wsum;
    out_w[token * 2 + 1]   = b2 * wsum;

    atomicAdd(&s_cnt[i1], 1.0f);   // ds_add_f32
    atomicAdd(&s_cnt[i2], 1.0f);
  }
  __syncthreads();

  // ---- per-expert block reduction -> global accumulators --------------------
  if (tid < NEXP) {
    float s = 0.0f;
    for (int r = 0; r < BM; ++r) s += s_logits[r * NEXP + tid];
    atomicAdd(&g_imp[tid], s);
    atomicAdd(&g_cnt[tid], s_cnt[tid]);
  }
}

// ---------------- load-balance loss ----------------
__global__ void Router_loss_kernel(const float* __restrict__ cnt,
                                   const float* __restrict__ imp,
                                   float* __restrict__ loss, float nTok) {
  if (threadIdx.x == 0) {
    float s = 0.0f;
    for (int e = 0; e < NEXP; ++e)
      s += (cnt[e] / (nTok * 2.0f)) * (imp[e] / nTok);
    *loss = (float)NEXP * s;
  }
}

extern "C" void kernel_launch(void* const* d_in, const int* in_sizes, int n_in,
                              void* d_out, int out_size, void* d_ws, size_t ws_size,
                              hipStream_t stream) {
  const float* x = (const float*)d_in[0];
  const float* W = (const float*)d_in[1];
  const float* b = (const float*)d_in[2];
  const int N = in_sizes[0] / D_MODEL;   // 32768 tokens

  // workspace: [ Wbf16 padded pitch : 64*2064*2 B ][ g_cnt : 64 f32 ][ g_imp : 64 f32 ]
  __bf16* Wb    = (__bf16*)d_ws;
  float*  accs  = (float*)((char*)d_ws + (size_t)LDS_W_BYTES);
  float*  g_cnt = accs;
  float*  g_imp = accs + NEXP;

  // output: idx[N*2] | weights[N*2] | loss[1] | logits[N*64]
  float* out        = (float*)d_out;
  float* out_idx    = out;
  float* out_w      = out + (size_t)N * 2;
  float* out_loss   = out + (size_t)N * 4;
  float* out_logits = out + (size_t)N * 4 + 1;

  Router_cvtw_kernel<<<(NEXP * D_MODEL + 255) / 256, 256, 0, stream>>>(
      W, Wb, NEXP * D_MODEL);
  Router_zero_kernel<<<1, 128, 0, stream>>>(accs);
  Router_main_kernel<<<N / BM, 256, LDS_TOTAL, stream>>>(
      x, Wb, b, out_idx, out_w, out_logits, g_cnt, g_imp);
  Router_loss_kernel<<<1, 32, 0, stream>>>(g_cnt, g_imp, out_loss, (float)N);
}
